// SRNN_13950053777986
// MI455X (gfx1250) — matrix operand order repaired and checked by reading, not compiled, not run on hardware
//
#include <hip/hip_runtime.h>
#include <hip/hip_bf16.h>
#include <math.h>

// ---------------------------------------------------------------------------
// SRNN on MI455X (gfx1250).
// Dominant cost: spatial edge GRU over 65536 edges/step ->
//   GEMM1 [65536 x 64] x [64 x 384]   (input gates)
//   GEMM2 [65536 x 128] x [128 x 384] (hidden gates)
// ~97 GFLOP total; He state (33.5MB) is L2-resident (192MB L2) => compute
// bound => v_wmma_f32_16x16x32_f16 mixed precision path.
// k2 layout: 512 threads = 16 waves = 2 M-subtiles x 8 column-triplets.
// Each wave owns output channels {c, c+128, c+256} so the whole GRU gate
// combine is done in registers (no gate LDS, no barrier). He rows staged
// once in LDS to kill the per-wave duplicate global loads.
// ---------------------------------------------------------------------------

typedef _Float16 v16h __attribute__((ext_vector_type(16)));
typedef float    v8f  __attribute__((ext_vector_type(8)));
typedef float    v4f  __attribute__((ext_vector_type(4)));

#define TT   10
#define NN   256
#define ERNN 128

#define HPAD 132   // padded LDS row stride (floats) -> conflict-free strided reads

__device__ __forceinline__ float sigf(float x) { return 1.0f / (1.0f + __expf(-x)); }

// ---------------------------------------------------------------------------
// K0: convert spatial GRU weights f32 -> f16 (row-major [384][K])
// ---------------------------------------------------------------------------
__global__ __launch_bounds__(256) void k0_cvt(const float* __restrict__ s_wih,
                                              const float* __restrict__ s_whh,
                                              _Float16* __restrict__ wih16,
                                              _Float16* __restrict__ whh16) {
  int i = blockIdx.x * 256 + threadIdx.x;
  if (i < 384 * 64)  wih16[i] = (_Float16)s_wih[i];
  if (i < 384 * 128) whh16[i] = (_Float16)s_whh[i];
}

// ---------------------------------------------------------------------------
// K1: temporal GRU on diagonal edges (reads OLD He[i,i]) -> ht_buf[256][128]
// ---------------------------------------------------------------------------
__global__ __launch_bounds__(128) void k1_diag(
    const float* __restrict__ edges_t, const float* __restrict__ He,
    const float* __restrict__ tew, const float* __restrict__ teb,
    const float* __restrict__ twih, const float* __restrict__ twhh,
    const float* __restrict__ tbih, const float* __restrict__ tbhh,
    float* __restrict__ ht_buf) {
  int i = blockIdx.x, c = threadIdx.x;
  __shared__ float emb[64];
  __shared__ float hsh[128];
  int e = i * NN + i;
  const float* hrow = He + (size_t)e * ERNN;
  hsh[c] = hrow[c];
  if (c < 64) {
    float f0 = edges_t[2 * e], f1 = edges_t[2 * e + 1];
    emb[c] = fmaxf(fmaf(tew[2 * c], f0, fmaf(tew[2 * c + 1], f1, teb[c])), 0.0f);
  }
  __syncthreads();
  float gr = tbih[c], gz = tbih[128 + c], gn = tbih[256 + c];
  for (int k = 0; k < 64; ++k) {
    float xk = emb[k];
    gr = fmaf(xk, twih[c * 64 + k], gr);
    gz = fmaf(xk, twih[(128 + c) * 64 + k], gz);
    gn = fmaf(xk, twih[(256 + c) * 64 + k], gn);
  }
  float hr = tbhh[c], hz = tbhh[128 + c], hn = tbhh[256 + c];
  for (int k = 0; k < 128; ++k) {
    float hk = hsh[k];
    hr = fmaf(hk, twhh[c * 128 + k], hr);
    hz = fmaf(hk, twhh[(128 + c) * 128 + k], hz);
    hn = fmaf(hk, twhh[(256 + c) * 128 + k], hn);
  }
  float r = sigf(gr + hr);
  float z = sigf(gz + hz);
  float n = tanhf(gn + r * hn);
  ht_buf[i * 128 + c] = (1.0f - z) * n + z * hsh[c];
}

// ---------------------------------------------------------------------------
// K2: spatial GRU on all 65536 edges via WMMA.
// Block = 32 edges, 512 threads (16 waves). wave = msub(2) x colgroup(8).
// Wave w owns channels 16*(w&7) .. +15 for all three gates (r,z,n) so the
// GRU combine is register-local. 18 v_wmma per wave. Diagonal stores
// skipped (K3 fixes them from the temporal GRU).
// ---------------------------------------------------------------------------
__global__ __launch_bounds__(512) void k2_spatial(
    const float* __restrict__ edges_t, const unsigned char* __restrict__ em_t,
    const float* __restrict__ enc_w, const float* __restrict__ enc_b,
    const _Float16* __restrict__ wih16, const _Float16* __restrict__ whh16,
    const float* __restrict__ bih, const float* __restrict__ bhh,
    float* __restrict__ He) {
  __shared__ float sw[128];
  __shared__ float sb[64];
  __shared__ float hstage[32 * HPAD];  // 32 He rows, padded stride
  int tid = threadIdx.x;
  if (tid < 128) sw[tid] = enc_w[tid];
  if (tid < 64)  sb[tid] = enc_b[tid];

  int eBase = blockIdx.x * 32;
  // ---- stage 32 He rows (old hidden state) into LDS once per block
  {
    const float* src = He + (size_t)eBase * ERNN;
    for (int idx = tid; idx < 32 * 32; idx += 512) {
      int row = idx >> 5, p = (idx & 31) * 4;
      *(v4f*)(hstage + row * HPAD + p) = *(const v4f*)(src + row * 128 + p);
    }
  }
  __syncthreads();

  int wave = tid >> 5, lane = tid & 31;
  int msub = wave >> 3;          // 0..1  : which 16-edge subtile
  int cg   = (wave & 7) * 16;    // 0..112: channel group base
  int m = lane & 15, hi = lane >> 4;
  int row16 = msub * 16 + m;     // local A row 0..31
  int e = eBase + row16;

  // ---- A fragment: x-embedding (16x64 f16) = relu(enc_w @ feat + b)
  float f0 = edges_t[2 * e], f1 = edges_t[2 * e + 1];
  v16h ax[2];
#pragma unroll
  for (int q = 0; q < 2; ++q) {
#pragma unroll
    for (int j = 0; j < 16; ++j) {
      int k = 32 * q + hi * 16 + j;  // contiguous k = 16*hi + j (matches B)
      float v = fmaf(sw[2 * k], f0, fmaf(sw[2 * k + 1], f1, sb[k]));
      ax[q][j] = (_Float16)fmaxf(v, 0.0f);
    }
  }
  // ---- A fragment: hidden state (16x128) from LDS stage, f32 -> f16
  const float* hrow = hstage + row16 * HPAD;
  v16h ah[4];
#pragma unroll
  for (int q = 0; q < 4; ++q) {
#pragma unroll
    for (int r2 = 0; r2 < 2; ++r2) {
      int k0 = 32 * q + 16 * hi + 8 * r2;
      v4f a = *(const v4f*)(hrow + k0);
      v4f b = *(const v4f*)(hrow + k0 + 4);
#pragma unroll
      for (int j = 0; j < 4; ++j) {
        ah[q][r2 * 8 + j]     = (_Float16)a[j];
        ah[q][r2 * 8 + 4 + j] = (_Float16)b[j];
      }
    }
  }

  // ---- 3 gate tiles (r,z,n) for this wave's 16 channels
  v8f gi[3], gh[3];
#pragma unroll
  for (int g = 0; g < 3; ++g) {
    int wr = cg + 128 * g + m;  // W row == output channel (B = W^T)
    const _Float16* w0 = wih16 + (size_t)wr * 64;
    v8f acc = {};
#pragma unroll
    for (int q = 0; q < 2; ++q) {
      v16h bf = *(const v16h*)(w0 + 32 * q + hi * 16);  // contiguous k = 16*hi + s, same as A
      acc = __builtin_amdgcn_wmma_f32_16x16x32_f16(false, ax[q], false, bf,
                                                   (short)0, acc, false, false);
    }
    gi[g] = acc;
    const _Float16* w1 = whh16 + (size_t)wr * 128;
    v8f acch = {};
#pragma unroll
    for (int q = 0; q < 4; ++q) {
      v16h bf = *(const v16h*)(w1 + 32 * q + hi * 16);
      acch = __builtin_amdgcn_wmma_f32_16x16x32_f16(false, ah[q], false, bf,
                                                    (short)0, acch, false, false);
    }
    gh[g] = acch;
  }

  // ---- register-local GRU combine + masked in-place state update
  int c = cg + m;  // D layout: col = lane&15 -> channel cg+m
  float bi_r = bih[c],       bh_r = bhh[c];
  float bi_z = bih[128 + c], bh_z = bhh[128 + c];
  float bi_n = bih[256 + c], bh_n = bhh[256 + c];
#pragma unroll
  for (int r = 0; r < 8; ++r) {
    int row = msub * 16 + r + 8 * hi;  // D layout: row = r + 8*hi
    int ee = eBase + row;
    float hold = hstage[row * HPAD + c];
    float rs = sigf(gi[0][r] + gh[0][r] + bi_r + bh_r);
    float zs = sigf(gi[1][r] + gh[1][r] + bi_z + bh_z);
    float nn = tanhf(gi[2][r] + bi_n + rs * (gh[2][r] + bh_n));
    float hnew = (1.0f - zs) * nn + zs * hold;
    int ii = ee >> 8, jj = ee & 255;
    if (jj != ii && em_t[ee]) He[(size_t)ee * ERNN + c] = hnew;
  }
}

// ---------------------------------------------------------------------------
// K3: store masked diagonal (temporal state) + aggregate off-diag edge states
// ---------------------------------------------------------------------------
__global__ __launch_bounds__(128) void k3_agg(
    const unsigned char* __restrict__ em_t, const float* __restrict__ ht_buf,
    float* __restrict__ He, float* __restrict__ hsp) {
  int i = blockIdx.x, c = threadIdx.x;
  int ed = i * NN + i;
  if (em_t[ed]) He[(size_t)ed * ERNN + c] = ht_buf[i * 128 + c];
  float acc = 0.0f;
  for (int j = 0; j < NN; ++j) {
    if (j == i) continue;
    if (em_t[i * NN + j]) acc += He[((size_t)i * NN + j) * ERNN + c];
  }
  hsp[i * 128 + c] = acc;
}

// ---------------------------------------------------------------------------
// K4: node GRU + decoder. Block = node i (128 threads)
// ---------------------------------------------------------------------------
__global__ __launch_bounds__(128) void k4_node(
    const float* __restrict__ nodes_t, const unsigned char* __restrict__ nm_t,
    const float* __restrict__ He, const float* __restrict__ hsp,
    const float* __restrict__ new_w, const float* __restrict__ new_b,
    const float* __restrict__ nhw, const float* __restrict__ nhb,
    const float* __restrict__ nwih, const float* __restrict__ nwhh,
    const float* __restrict__ nbih, const float* __restrict__ nbhh,
    const float* __restrict__ decw, const float* __restrict__ decb,
    float* __restrict__ Hn, float* __restrict__ out_t) {
  int i = blockIdx.x, c = threadIdx.x;
  __shared__ float hoth[256];    // [He_diag(128) ; h_spatial(128)]
  __shared__ float xcat[128];    // [enc_x(64) ; enc_h(64)]
  __shared__ float hold_s[128];
  __shared__ float hn_s[128];
  hoth[c]       = He[((size_t)i * (NN + 1)) * ERNN + c];
  hoth[128 + c] = hsp[i * 128 + c];
  hold_s[c]     = Hn[i * 128 + c];
  __syncthreads();
  if (c < 64) {
    float x0 = nodes_t[2 * i], x1 = nodes_t[2 * i + 1];
    xcat[c] = fmaxf(fmaf(new_w[2 * c], x0, fmaf(new_w[2 * c + 1], x1, new_b[c])), 0.0f);
  } else {
    int cc = c - 64;
    float a = nhb[cc];
    for (int k = 0; k < 256; ++k) a = fmaf(hoth[k], nhw[cc * 256 + k], a);
    xcat[c] = fmaxf(a, 0.0f);
  }
  __syncthreads();
  float gr = nbih[c], gz = nbih[128 + c], gn = nbih[256 + c];
  for (int k = 0; k < 128; ++k) {
    float xk = xcat[k];
    gr = fmaf(xk, nwih[c * 128 + k], gr);
    gz = fmaf(xk, nwih[(128 + c) * 128 + k], gz);
    gn = fmaf(xk, nwih[(256 + c) * 128 + k], gn);
  }
  float hr = nbhh[c], hz = nbhh[128 + c], hnv = nbhh[256 + c];
  for (int k = 0; k < 128; ++k) {
    float hk = hold_s[k];
    hr  = fmaf(hk, nwhh[c * 128 + k], hr);
    hz  = fmaf(hk, nwhh[(128 + c) * 128 + k], hz);
    hnv = fmaf(hk, nwhh[(256 + c) * 128 + k], hnv);
  }
  float r = sigf(gr + hr);
  float z = sigf(gz + hz);
  float n = tanhf(gn + r * hnv);
  float hnew = (1.0f - z) * n + z * hold_s[c];
  bool nm = nm_t[i] != 0;
  float hval = nm ? hnew : hold_s[c];
  Hn[i * 128 + c] = hval;
  hn_s[c] = hval;
  __syncthreads();
  if (c < 5) {
    float o = decb[c];
    for (int k = 0; k < 128; ++k) o = fmaf(hn_s[k], decw[c * 128 + k], o);
    out_t[i * 5 + c] = nm ? o : 0.0f;
  }
}

// ---------------------------------------------------------------------------
extern "C" void kernel_launch(void* const* d_in, const int* in_sizes, int n_in,
                              void* d_out, int out_size, void* d_ws, size_t ws_size,
                              hipStream_t stream) {
  (void)in_sizes; (void)n_in; (void)out_size; (void)ws_size;
  const float* nodes   = (const float*)d_in[0];
  const float* edges   = (const float*)d_in[1];
  const unsigned char* node_mask = (const unsigned char*)d_in[2];
  const unsigned char* edge_mask = (const unsigned char*)d_in[3];
  const float* t_enc_w = (const float*)d_in[4];
  const float* t_enc_b = (const float*)d_in[5];
  const float* t_wih   = (const float*)d_in[6];
  const float* t_whh   = (const float*)d_in[7];
  const float* t_bih   = (const float*)d_in[8];
  const float* t_bhh   = (const float*)d_in[9];
  const float* s_enc_w = (const float*)d_in[10];
  const float* s_enc_b = (const float*)d_in[11];
  const float* s_wih   = (const float*)d_in[12];
  const float* s_whh   = (const float*)d_in[13];
  const float* s_bih   = (const float*)d_in[14];
  const float* s_bhh   = (const float*)d_in[15];
  const float* n_enc_w = (const float*)d_in[16];
  const float* n_enc_b = (const float*)d_in[17];
  const float* n_henc_w= (const float*)d_in[18];
  const float* n_henc_b= (const float*)d_in[19];
  const float* n_wih   = (const float*)d_in[20];
  const float* n_whh   = (const float*)d_in[21];
  const float* n_bih   = (const float*)d_in[22];
  const float* n_bhh   = (const float*)d_in[23];
  const float* dec_w   = (const float*)d_in[24];
  const float* dec_b   = (const float*)d_in[25];
  float* out = (float*)d_out;

  // Workspace layout (floats): He | ht | hsp | Hn | wih16 | whh16  (~34.2 MB)
  float* He  = (float*)d_ws;                 // 256*256*128
  float* ht  = He  + (size_t)NN * NN * ERNN; // 256*128
  float* hsp = ht  + NN * 128;
  float* Hn  = hsp + NN * 128;
  _Float16* wih16 = (_Float16*)(Hn + NN * 128);  // 384*64
  _Float16* whh16 = wih16 + 384 * 64;            // 384*128

  // zero recurrent state (He, ht, hsp, Hn are contiguous)
  hipMemsetAsync(He, 0, ((size_t)NN * NN * ERNN + 3 * NN * 128) * sizeof(float), stream);
  k0_cvt<<<192, 256, 0, stream>>>(s_wih, s_whh, wih16, whh16);

  for (int t = 0; t < TT; ++t) {
    const float* ed_t = edges + (size_t)t * NN * NN * 2;
    const unsigned char* em_t = edge_mask + (size_t)t * NN * NN;
    k1_diag<<<NN, 128, 0, stream>>>(ed_t, He, t_enc_w, t_enc_b, t_wih, t_whh,
                                    t_bih, t_bhh, ht);
    k2_spatial<<<NN * NN / 32, 512, 0, stream>>>(ed_t, em_t, s_enc_w, s_enc_b,
                                                 wih16, whh16, s_bih, s_bhh, He);
    k3_agg<<<NN, 128, 0, stream>>>(em_t, ht, He, hsp);
    k4_node<<<NN, 128, 0, stream>>>(nodes + (size_t)t * NN * 2,
                                    node_mask + (size_t)t * NN, He, hsp,
                                    n_enc_w, n_enc_b, n_henc_w, n_henc_b,
                                    n_wih, n_whh, n_bih, n_bhh, dec_w, dec_b,
                                    Hn, out + (size_t)t * NN * 5);
  }
}
